// Gemma3nAttention_5308579577900
// MI455X (gfx1250) — matrix operand order repaired and checked by reading, not compiled, not run on hardware
//
#include <hip/hip_runtime.h>
#include <hip/hip_bf16.h>
#include <cstdint>

#define B_   2
#define S_   2048
#define HIDN 2048
#define NH_  8
#define NKV_ 2
#define HD_  256
#define SW_  512

typedef __attribute__((ext_vector_type(16))) __bf16 bf16x16;
typedef __attribute__((ext_vector_type(8)))  float  f32x8;
typedef __attribute__((__vector_size__(16))) int    v4i;

union FragU { bf16x16 v; uint4 q[2]; };

// ---- CDNA5 async global->LDS path (guarded; falls back to vector copy) ----
#if defined(__gfx1250__) && __has_builtin(__builtin_amdgcn_global_load_async_to_lds_b128)
#define HAVE_ASYNC_LDS 1
__device__ __forceinline__ void async_cp16(unsigned short* lds, const unsigned short* g){
  __builtin_amdgcn_global_load_async_to_lds_b128(
      (__attribute__((address_space(1))) v4i*)(void*)const_cast<unsigned short*>(g),
      (__attribute__((address_space(3))) v4i*)(void*)lds, 0, 0);
}
__device__ __forceinline__ void wait_async0(){
#if __has_builtin(__builtin_amdgcn_s_wait_asynccnt)
  __builtin_amdgcn_s_wait_asynccnt(0);
#else
  asm volatile("s_wait_asynccnt 0x0" ::: "memory");
#endif
}
#else
#define HAVE_ASYNC_LDS 0
__device__ __forceinline__ void wait_async0(){}
#endif

__device__ __forceinline__ unsigned short f2bf(float f){
  unsigned int u = __float_as_uint(f);
  u += 0x7FFFu + ((u >> 16) & 1u);          // round-to-nearest-even
  return (unsigned short)(u >> 16);
}

__device__ __forceinline__ f32x8 wmma_bf16(bf16x16 a, bf16x16 b, f32x8 c){
  return __builtin_amdgcn_wmma_f32_16x16x32_bf16(false, a, false, b, (short)0, c, false, false);
}

// A fragment: LDS tile stored [M][K] (K contiguous). Lane L: m=row0+(L&15),
// 16 bf16 at K = g*8+{0..7} and 16+g*8+{0..7}, g = L>>4.
__device__ __forceinline__ bf16x16 frag_a(const unsigned short* base, int row0,
                                          int rs, int kofs, int lane){
  const unsigned short* p = base + (row0 + (lane & 15)) * rs + kofs + ((lane >> 4) << 3);
  FragU f;
  f.q[0] = *(const uint4*)p;
  f.q[1] = *(const uint4*)(p + 16);
  return f.v;
}

// B fragment: LDS tile stored [N][K] (K contiguous). Lane L: n=col0+(L&15),
// 16 bf16 at K = g*8+{0..7} and 16+g*8+{0..7}, g = L>>4.
__device__ __forceinline__ bf16x16 frag_b(const unsigned short* base, int col0,
                                          int rs, int kofs, int lane){
  const unsigned short* p = base + (col0 + (lane & 15)) * rs + kofs + ((lane >> 4) << 3);
  FragU f;
  f.q[0] = *(const uint4*)p;
  f.q[1] = *(const uint4*)(p + 16);
  return f.v;
}

__device__ __forceinline__ void st4bf(unsigned short* p, float4 f){
  union { unsigned short u[4]; uint2 v2; } t;
  t.u[0] = f2bf(f.x); t.u[1] = f2bf(f.y); t.u[2] = f2bf(f.z); t.u[3] = f2bf(f.w);
  *(uint2*)p = t.v2;
}

// ---------------------------------------------------------------------------
// Kernel 1: fused QKV projection + per-head RMS norm + RoPE, bf16 outputs.
// grid = (B*S/32, 12): blockIdx.y selects q head 0..7, k head 8..9, v head 10..11
// ---------------------------------------------------------------------------
__global__ __launch_bounds__(256) void qkv_kernel(
    const float* __restrict__ hidden, const float* __restrict__ Wq,
    const float* __restrict__ Wk, const float* __restrict__ Wv,
    const float* __restrict__ qnw, const float* __restrict__ knw,
    const int* __restrict__ positions,
    unsigned short* __restrict__ qb, unsigned short* __restrict__ kbuf,
    unsigned short* __restrict__ vbuf)
{
  __shared__ __align__(16) unsigned char smem[(32*64 + 256*64) * 2];
  unsigned short* As = (unsigned short*)smem;   // [32][64]
  unsigned short* Bs = As + 32*64;              // [256][64]

  const int tid  = threadIdx.x;
  const int w    = tid >> 5, lane = tid & 31;
  const int hsel = blockIdx.y;
  const int t0   = blockIdx.x * 32;

  const float* W; const float* nw; int rope;
  if (hsel < NH_)            { W = Wq + (size_t)hsel*HD_*HIDN;              nw = qnw;     rope = 1; }
  else if (hsel < NH_+NKV_)  { W = Wk + (size_t)(hsel-NH_)*HD_*HIDN;        nw = knw;     rope = 1; }
  else                       { W = Wv + (size_t)(hsel-NH_-NKV_)*HD_*HIDN;   nw = nullptr; rope = 0; }

  const int mrow = (w >> 2) * 16;   // 2 M-tiles of 16
  const int nb0  = (w & 3) * 64;    // 4 N-groups of 64
  f32x8 acc[4] = {};

  for (int kt = 0; kt < HIDN/64; ++kt){
    const int k0 = kt * 64;
    // stage A: 32x64 f32 -> bf16
    #pragma unroll
    for (int j = 0; j < 2; ++j){
      int slot = tid + 256*j;
      int m = slot >> 4, kq = (slot & 15) * 4;
      float4 f = *(const float4*)(hidden + (size_t)(t0+m)*HIDN + k0 + kq);
      st4bf(As + m*64 + kq, f);
    }
    // stage B: 256x64 f32 -> bf16 (W rows are K-contiguous already)
    #pragma unroll
    for (int j = 0; j < 16; ++j){
      int slot = tid + 256*j;
      int n = slot >> 4, kq = (slot & 15) * 4;
      float4 f = *(const float4*)(W + (size_t)n*HIDN + k0 + kq);
      st4bf(Bs + n*64 + kq, f);
    }
    __syncthreads();
    #pragma unroll
    for (int ks = 0; ks < 2; ++ks){
      bf16x16 a = frag_a(As, mrow, 64, ks*32, lane);
      bf16x16 bm[4];
      #pragma unroll
      for (int t = 0; t < 4; ++t) bm[t] = frag_b(Bs, nb0 + t*16, 64, ks*32, lane);
      #pragma unroll
      for (int t = 0; t < 4; ++t) acc[t] = wmma_bf16(a, bm[t], acc[t]);
    }
    __syncthreads();
  }

  // ---- epilogue: RMS over the 256-dim head row, then RoPE ----
  float* nbuf = (float*)smem;       // [32][256]
  float* rstd = nbuf + 32*256;      // [32]
  #pragma unroll
  for (int t = 0; t < 4; ++t)
    #pragma unroll
    for (int r = 0; r < 8; ++r){
      int m = mrow + r + ((lane >> 4) << 3);
      int n = nb0 + t*16 + (lane & 15);
      nbuf[m*256 + n] = acc[t][r];
    }
  __syncthreads();
  if (tid < 32){
    const float* row = nbuf + tid*256;
    float s = 0.f;
    for (int d = 0; d < 256; ++d){ float x = row[d]; s += x * x; }
    rstd[tid] = rsqrtf(s * (1.0f/256.0f) + 1e-6f);
  }
  __syncthreads();
  for (int it = 0; it < 32; ++it){
    int e = it*256 + tid;
    int m = e >> 8, d = e & 255;
    float rs = rstd[m];
    float x = nbuf[m*256 + d] * rs * (nw ? nw[d] : 1.0f);
    float outv = x;
    if (rope){
      int i = d & 127;
      float inv = __expf(-(float)i * 0.0719557841560639f);  // ln(10000)/128
      float p   = (float)positions[t0 + m];
      float ang = p * inv;
      float sn, cs; __sincosf(ang, &sn, &cs);
      int dp = (d < 128) ? d + 128 : d - 128;
      float xp  = nbuf[m*256 + dp] * rs * (nw ? nw[dp] : 1.0f);
      float rot = (d < 128) ? -xp : xp;
      outv = x * cs + rot * sn;
    }
    int tg = t0 + m;
    int b = tg >> 11, s = tg & (S_ - 1);
    unsigned short bf = f2bf(outv);
    if (hsel < NH_)
      qb[(((size_t)(b*NH_ + hsel))*S_ + s)*HD_ + d] = bf;
    else if (hsel < NH_ + NKV_)
      kbuf[(((size_t)(b*NKV_ + (hsel - NH_)))*S_ + s)*HD_ + d] = bf;
    else
      vbuf[(((size_t)(b*NKV_ + (hsel - NH_ - NKV_)))*S_ + s)*HD_ + d] = bf;
  }
}

// ---------------------------------------------------------------------------
// Kernel 2: sliding-window flash attention (64 queries per WG, online softmax)
// grid = (S/64, NH, B)
// ---------------------------------------------------------------------------
__global__ __launch_bounds__(256) void attn_kernel(
    const unsigned short* __restrict__ qb, const unsigned short* __restrict__ kbuf,
    const unsigned short* __restrict__ vbuf, unsigned short* __restrict__ ab)
{
  __shared__ __align__(16) unsigned short Qs[64*256];  // 32 KB
  __shared__ __align__(16) unsigned short Ks[64*256];  // 32 KB
  __shared__ __align__(16) unsigned short Vt[256*64];  // 32 KB, [d][k]
  __shared__ __align__(16) float          Sb[64*64];   // 16 KB
  __shared__ __align__(16) unsigned short Pb[64*64];   // 8 KB
  __shared__ float mbuf[64], lbuf[64], abuf[64];

  const int tid = threadIdx.x, w = tid >> 5, lane = tid & 31;
  const int q0 = blockIdx.x * 64;
  const int h = blockIdx.y, b = blockIdx.z;
  const int kvh = h / (NH_ / NKV_);   // jnp.repeat grouping

  const unsigned short* qsrc  = qb   + (((size_t)(b*NH_  + h  ))*S_ + q0) * HD_;
  const unsigned short* kbase = kbuf + ((size_t)(b*NKV_ + kvh))*S_ * HD_;
  const unsigned short* vbase = vbuf + ((size_t)(b*NKV_ + kvh))*S_ * HD_;

#if HAVE_ASYNC_LDS
  for (int i = tid; i < 64*256/8; i += 256)
    async_cp16(Qs + i*8, qsrc + i*8);
#else
  for (int i = tid; i < 64*256/8; i += 256)
    ((uint4*)Qs)[i] = ((const uint4*)qsrc)[i];
#endif
  if (tid < 64){ mbuf[tid] = -3.4028235e38f; lbuf[tid] = 0.f; }

  const int qi = w & 3;    // q sub-tile (rows qi*16..qi*16+15)
  const int dh = w >> 2;   // d half for O (dh*128), also k-tile pair for scores
  f32x8 o[8] = {};

  int lo = q0 - (SW_ - 1); if (lo < 0) lo = 0;
  const int kb_lo = lo >> 6, kb_hi = q0 >> 6;

  for (int kblk = kb_lo; kblk <= kb_hi; ++kblk){
    const int kk0 = kblk * 64;
    __syncthreads();   // Qs ready / previous Ks,Vt consumed
    const unsigned short* ksrc = kbase + (size_t)kk0 * HD_;
    const unsigned short* vsrc = vbase + (size_t)kk0 * HD_;
#if HAVE_ASYNC_LDS
    for (int i = tid; i < 64*256/8; i += 256)
      async_cp16(Ks + i*8, ksrc + i*8);
#else
    for (int i = tid; i < 64*256/8; i += 256)
      ((uint4*)Ks)[i] = ((const uint4*)ksrc)[i];
#endif
    for (int i = tid; i < 64*256; i += 256){   // transpose V -> [d][k]
      int kk = i >> 8, d = i & 255;
      Vt[d*64 + kk] = vsrc[i];
    }
    if (kblk < kb_hi){  // prefetch next key/value block into L2 (global_prefetch_b8)
      __builtin_prefetch(kbase + (size_t)(kk0 + 64) * HD_ + (tid << 6), 0, 1);
      __builtin_prefetch(vbase + (size_t)(kk0 + 64) * HD_ + (tid << 6), 0, 1);
    }
    wait_async0();
    __syncthreads();

    // S = Q . K^T  (each wave: 16q x 32k, K=256)
    f32x8 sacc[2] = {};
    #pragma unroll
    for (int ks = 0; ks < 8; ++ks){
      bf16x16 a  = frag_a(Qs, qi*16, 256, ks*32, lane);
      bf16x16 b0 = frag_b(Ks, (dh*2 + 0)*16, 256, ks*32, lane);
      bf16x16 b1 = frag_b(Ks, (dh*2 + 1)*16, 256, ks*32, lane);
      sacc[0] = wmma_bf16(a, b0, sacc[0]);
      sacc[1] = wmma_bf16(a, b1, sacc[1]);
    }
    #pragma unroll
    for (int t = 0; t < 2; ++t)
      #pragma unroll
      for (int r = 0; r < 8; ++r){
        int qm = qi*16 + r + ((lane >> 4) << 3);
        int kn = (dh*2 + t)*16 + (lane & 15);
        int qpos = q0 + qm, kpos = kk0 + kn;
        bool ok = (kpos <= qpos) && (kpos > qpos - SW_);
        Sb[qm*64 + kn] = ok ? sacc[t][r] : -3.4028235e38f;
      }
    __syncthreads();

    // online softmax row update (f32 stats, bf16 P)
    if (tid < 64){
      float m_old = mbuf[tid];
      const float* srow = Sb + tid*64;
      float rmax = -3.4028235e38f;
      for (int j = 0; j < 64; ++j) rmax = fmaxf(rmax, srow[j]);
      float m_new = fmaxf(m_old, rmax);
      float alpha = __expf(m_old - m_new);
      float rsum = 0.f;
      unsigned short* prow = Pb + tid*64;
      for (int j = 0; j < 64; ++j){
        float e = __expf(srow[j] - m_new);
        rsum += e;
        prow[j] = f2bf(e);
      }
      mbuf[tid] = m_new; abuf[tid] = alpha;
      lbuf[tid] = alpha * lbuf[tid] + rsum;
    }
    __syncthreads();

    // rescale O and accumulate P . V  (each wave: 16q x 128d, K=64)
    float al[8];
    #pragma unroll
    for (int r = 0; r < 8; ++r) al[r] = abuf[qi*16 + r + ((lane >> 4) << 3)];
    #pragma unroll
    for (int t = 0; t < 8; ++t)
      #pragma unroll
      for (int r = 0; r < 8; ++r) o[t][r] *= al[r];
    #pragma unroll
    for (int ks = 0; ks < 2; ++ks){
      bf16x16 a = frag_a(Pb, qi*16, 64, ks*32, lane);
      bf16x16 bm[8];
      #pragma unroll
      for (int t = 0; t < 8; ++t) bm[t] = frag_b(Vt, dh*128 + t*16, 64, ks*32, lane);
      #pragma unroll
      for (int t = 0; t < 8; ++t) o[t] = wmma_bf16(a, bm[t], o[t]);
    }
  }
  __syncthreads();

  float linv[8];
  #pragma unroll
  for (int r = 0; r < 8; ++r) linv[r] = 1.0f / lbuf[qi*16 + r + ((lane >> 4) << 3)];
  #pragma unroll
  for (int t = 0; t < 8; ++t)
    #pragma unroll
    for (int r = 0; r < 8; ++r){
      int qm = qi*16 + r + ((lane >> 4) << 3);
      int d  = dh*128 + t*16 + (lane & 15);
      ab[((size_t)(b*S_ + q0 + qm))*(NH_*HD_) + h*HD_ + d] = f2bf(o[t][r] * linv[r]);
    }
}

// ---------------------------------------------------------------------------
// Kernel 3: output projection, f32 result.  grid = (B*S/32, HID/256)
// ---------------------------------------------------------------------------
__global__ __launch_bounds__(256) void oproj_kernel(
    const unsigned short* __restrict__ ab, const float* __restrict__ Wo,
    float* __restrict__ out)
{
  __shared__ __align__(16) unsigned short As[32*64];
  __shared__ __align__(16) unsigned short Bs[256*64];
  const int tid = threadIdx.x, w = tid >> 5, lane = tid & 31;
  const int t0 = blockIdx.x * 32;
  const int n0 = blockIdx.y * 256;
  const int mrow = (w >> 2) * 16, nb0 = (w & 3) * 64;
  f32x8 acc[4] = {};

  for (int kt = 0; kt < (NH_*HD_)/64; ++kt){
    const int k0 = kt * 64;
    { // A already bf16: straight b128 copy
      int m = tid >> 3, c = tid & 7;
      *(uint4*)(As + m*64 + c*8) =
          *(const uint4*)(ab + (size_t)(t0+m)*(NH_*HD_) + k0 + c*8);
    }
    #pragma unroll
    for (int j = 0; j < 16; ++j){
      int slot = tid + 256*j;
      int n = slot >> 4, kq = (slot & 15) * 4;
      float4 f = *(const float4*)(Wo + (size_t)(n0+n)*(NH_*HD_) + k0 + kq);
      st4bf(Bs + n*64 + kq, f);
    }
    __syncthreads();
    #pragma unroll
    for (int ks = 0; ks < 2; ++ks){
      bf16x16 a = frag_a(As, mrow, 64, ks*32, lane);
      bf16x16 bm[4];
      #pragma unroll
      for (int t = 0; t < 4; ++t) bm[t] = frag_b(Bs, nb0 + t*16, 64, ks*32, lane);
      #pragma unroll
      for (int t = 0; t < 4; ++t) acc[t] = wmma_bf16(a, bm[t], acc[t]);
    }
    __syncthreads();
  }
  #pragma unroll
  for (int t = 0; t < 4; ++t)
    #pragma unroll
    for (int r = 0; r < 8; ++r){
      int m = t0 + mrow + r + ((lane >> 4) << 3);
      int n = n0 + nb0 + t*16 + (lane & 15);
      out[(size_t)m*HIDN + n] = acc[t][r];
    }
}

// ---------------------------------------------------------------------------
extern "C" void kernel_launch(void* const* d_in, const int* in_sizes, int n_in,
                              void* d_out, int out_size, void* d_ws, size_t ws_size,
                              hipStream_t stream)
{
  (void)in_sizes; (void)n_in; (void)out_size; (void)ws_size;
  const int*   positions = (const int*)d_in[0];
  const float* hidden    = (const float*)d_in[1];
  const float* Wq        = (const float*)d_in[2];
  const float* Wk        = (const float*)d_in[3];
  const float* Wv        = (const float*)d_in[4];
  const float* Wo        = (const float*)d_in[5];
  const float* qnw       = (const float*)d_in[6];
  const float* knw       = (const float*)d_in[7];
  // d_in[8] = forward_batch scalar (unused)

  unsigned short* qb = (unsigned short*)d_ws;                 // [B][NH][S][HD] bf16
  unsigned short* kb = qb + (size_t)B_*NH_*S_*HD_;            // [B][NKV][S][HD]
  unsigned short* vb = kb + (size_t)B_*NKV_*S_*HD_;           // [B][NKV][S][HD]
  unsigned short* ab = vb + (size_t)B_*NKV_*S_*HD_;           // [B][S][NH*HD]
  float* out = (float*)d_out;

  qkv_kernel<<<dim3((B_*S_)/32, NH_ + 2*NKV_), 256, 0, stream>>>(
      hidden, Wq, Wk, Wv, qnw, knw, positions, qb, kb, vb);
  attn_kernel<<<dim3(S_/64, NH_, B_), 256, 0, stream>>>(qb, kb, vb, ab);
  oproj_kernel<<<dim3((B_*S_)/32, HIDN/256), 256, 0, stream>>>(ab, Wo, out);
}
